// Net_77137612636326
// MI455X (gfx1250) — matrix operand order repaired and checked by reading, not compiled, not run on hardware
//
#include <hip/hip_runtime.h>
#include <hip/hip_bf16.h>

#define NPTS 2048
#define NB   8
#define KNN  20
#define INVBN 0.99999500003749937f   // 1/sqrt(1+1e-5)
#define PI_F 3.14159265358979323f

typedef __attribute__((ext_vector_type(16))) _Float16 v16h;
typedef __attribute__((ext_vector_type(8)))  float    v8f;

__device__ __forceinline__ float lrelu_f(float x) { return x > 0.f ? x : 0.2f * x; }

// ---------------------------------------------------------------------------
// kNN: per-point top-20 of d = 2*inner - |xi|^2 - |xj|^2 (largest = nearest).
// Block = 128 points of one batch; candidate tiles staged candidate-major in
// LDS (channels padded to x4) so the inner product reads ds_load_b128.
// ---------------------------------------------------------------------------
template<int C>
__global__ void knn_kernel(const float* __restrict__ X, long xbst, int* __restrict__ idx)
{
    constexpr int CP = (C + 3) & ~3;
    __shared__ __align__(16) float sf[128 * CP];
    __shared__ float sxx[128];
    const int b   = blockIdx.y;
    const int tid = threadIdx.x;
    const int n   = blockIdx.x * 128 + tid;
    const float* Xb = X + (long)b * xbst;

    float f[CP];
    float myxx = 0.f;
#pragma unroll
    for (int c = 0; c < CP; ++c) {
        f[c] = (c < C) ? Xb[(long)c * NPTS + n] : 0.f;
        myxx += f[c] * f[c];
    }

    float tv[KNN]; int ti[KNN];
#pragma unroll
    for (int k = 0; k < KNN; ++k) { tv[k] = -3.4e38f; ti[k] = 0; }
    float minv = -3.4e38f; int minp = 0;

    for (int t = 0; t < NPTS / 128; ++t) {
        const int j0 = t * 128;
        float xx = 0.f;
#pragma unroll
        for (int c = 0; c < CP; ++c) {
            float v = (c < C) ? Xb[(long)c * NPTS + j0 + tid] : 0.f;
            sf[tid * CP + c] = v;
            xx += v * v;
        }
        sxx[tid] = xx;
        __syncthreads();
        for (int jj = 0; jj < 128; ++jj) {
            const float* sp = &sf[jj * CP];
            float inner = 0.f;
#pragma unroll
            for (int c4 = 0; c4 < CP; c4 += 4) {
                float4 s = *(const float4*)(sp + c4);
                inner += f[c4] * s.x + f[c4 + 1] * s.y + f[c4 + 2] * s.z + f[c4 + 3] * s.w;
            }
            float d = 2.f * inner - myxx - sxx[jj];
            if (d > minv) {
                tv[minp] = d; ti[minp] = j0 + jj;
                minv = tv[0]; minp = 0;
#pragma unroll
                for (int k = 1; k < KNN; ++k)
                    if (tv[k] < minv) { minv = tv[k]; minp = k; }
            }
        }
        __syncthreads();
    }
    int* op = idx + ((long)b * NPTS + n) * KNN;
#pragma unroll
    for (int k = 0; k < KNN; ++k) op[k] = ti[k];
}

// ---------------------------------------------------------------------------
// HOG features (18 ch) + copy xyz -> assembled 21-channel input XIN.
// ---------------------------------------------------------------------------
__global__ void hog_kernel(const float* __restrict__ src, const int* __restrict__ idx,
                           float* __restrict__ X)
{
    long t = (long)blockIdx.x * blockDim.x + threadIdx.x;
    if (t >= (long)NB * NPTS) return;
    int b = (int)(t / NPTS), n = (int)(t % NPTS);
    const float* S = src + (long)b * 3 * NPTS;
    float px = S[n], py = S[NPTS + n], pz = S[2 * NPTS + n];
    float ha[9], he[9];
#pragma unroll
    for (int i = 0; i < 9; ++i) { ha[i] = 0.f; he[i] = 0.f; }
    const int* ip = idx + ((long)b * NPTS + n) * KNN;
    for (int k = 0; k < KNN; ++k) {
        int j = ip[k];
        float dx = S[j] - px, dy = S[NPTS + j] - py, dz = S[2 * NPTS + j] - pz;
        float r2 = dx * dx + dy * dy;
        float mag = sqrtf(r2 + dz * dz + 1e-12f);
        float az = atan2f(dy, dx);
        float el = atan2f(dz, sqrtf(r2) + 1e-12f);
        int ba = (int)floorf((az + PI_F) / (2.f * PI_F) * 9.f);
        int be = (int)floorf((el + 0.5f * PI_F) / PI_F * 9.f);
        ba = min(max(ba, 0), 8); be = min(max(be, 0), 8);
        ha[ba] += mag; he[be] += mag;
    }
    float sa = 0.f, se = 0.f;
#pragma unroll
    for (int i = 0; i < 9; ++i) { sa += ha[i]; se += he[i]; }
    sa = 1.f / (sa + 1e-8f); se = 1.f / (se + 1e-8f);
    float* Xb = X + (long)b * 21 * NPTS;
    Xb[n] = px; Xb[NPTS + n] = py; Xb[2 * NPTS + n] = pz;
#pragma unroll
    for (int i = 0; i < 9; ++i) {
        Xb[(3 + i) * NPTS + n]  = ha[i] * sa;
        Xb[(12 + i) * NPTS + n] = he[i] * se;
    }
}

// ---------------------------------------------------------------------------
// Edge-conv weight split: W1 = w[:, :C], Wd = w[:, C:] - w[:, :C]
// ---------------------------------------------------------------------------
__global__ void prep_w_ec(const float* __restrict__ w, int O, int C,
                          float* __restrict__ W1, float* __restrict__ Wd)
{
    int t = blockIdx.x * blockDim.x + threadIdx.x;
    if (t >= O * C) return;
    int o = t / C, c = t % C;
    float a = w[o * 2 * C + c];
    W1[t] = a;
    Wd[t] = w[o * 2 * C + C + c] - a;
}

// ---------------------------------------------------------------------------
// Pack weights into per-lane A-fragment layout: Wp[rt][kt][lane][16 halves].
// All bounds checks / f32->f16 conversion happen here, once.
// ---------------------------------------------------------------------------
__global__ void pack_w(const float* __restrict__ W, int ldw, int O, int K, int KT,
                       _Float16* __restrict__ Wp)
{
    int rowTiles = (O + 15) >> 4;
    int t = blockIdx.x * blockDim.x + threadIdx.x;
    if (t >= rowTiles * KT * 32) return;
    int lane = t & 31;
    int kt   = (t >> 5) % KT;
    int rt   = (t >> 5) / KT;
    int grp  = lane >> 4;
    int mrow = rt * 16 + (lane & 15);
    v16h a;
#pragma unroll
    for (int e = 0; e < 16; ++e) {
        int kk = kt * 32 + 16 * grp + e;
        float v = (kk < K && mrow < O) ? W[(size_t)mrow * ldw + kk] : 0.f;
        a[e] = (_Float16)v;
    }
    *(v16h*)(Wp + (size_t)t * 16) = a;
}

// ---------------------------------------------------------------------------
// Pack activations k-contiguous per column: Xp[b][kt][n][32 halves].
// B-fragment for lane group g = contiguous 32B at offset g*16 halves.
// ---------------------------------------------------------------------------
__global__ void pack_x(const float* __restrict__ X, long xbst, int K, int KT,
                       _Float16* __restrict__ Xp)
{
    long t = (long)blockIdx.x * blockDim.x + threadIdx.x;
    if (t >= (long)NB * KT * NPTS) return;
    int n  = (int)(t % NPTS);
    int kt = (int)((t / NPTS) % KT);
    int b  = (int)(t / ((long)KT * NPTS));
    const float* xb = X + (long)b * xbst + (long)kt * 32 * NPTS + n;
    v16h lo, hi;
    if (kt * 32 + 32 <= K) {   // fast path: no bounds checks
#pragma unroll
        for (int e = 0; e < 16; ++e) lo[e] = (_Float16)xb[(long)e * NPTS];
#pragma unroll
        for (int e = 0; e < 16; ++e) hi[e] = (_Float16)xb[(long)(16 + e) * NPTS];
    } else {
#pragma unroll
        for (int e = 0; e < 16; ++e)
            lo[e] = (kt * 32 + e < K) ? (_Float16)xb[(long)e * NPTS] : (_Float16)0.f;
#pragma unroll
        for (int e = 0; e < 16; ++e)
            hi[e] = (kt * 32 + 16 + e < K) ? (_Float16)xb[(long)(16 + e) * NPTS] : (_Float16)0.f;
    }
    _Float16* out = Xp + t * 32;
    *(v16h*)out = lo;
    *(v16h*)(out + 16) = hi;
}

// ---------------------------------------------------------------------------
// WMMA GEMM on pre-packed fragments:  Y[b,o,n] = sum_k W[o,k]*X[b,k,n]
// Wave tile = 16(o) x 64(n); per K-step: 1 A-load + 4 B-loads (b128) + 4 WMMA.
// EPI: 0=raw  1=raw+abias  2=(+abias),bn,lrelu  3=bn,+addm,lrelu
// ---------------------------------------------------------------------------
template<int EPI>
__global__ void gemm_wmma(const _Float16* __restrict__ Wp,
                          const _Float16* __restrict__ Xp,
                          int O, int KT,
                          float* __restrict__ Y, long ybst,
                          const float* __restrict__ abias,
                          const float* __restrict__ g, const float* __restrict__ bb,
                          const float* __restrict__ addm, long addbst)
{
    const int rowTiles   = (O + 15) >> 4;
    const int totalWaves = rowTiles * (NB * (NPTS / 64));
    const int wid = blockIdx.x * (blockDim.x >> 5) + (threadIdx.x >> 5);
    if (wid >= totalWaves) return;                 // wave-uniform
    const int rt = wid % rowTiles;
    const int ct = wid / rowTiles;
    const int b  = ct >> 5;
    const int n0 = (ct & 31) * 64;
    const int lane = threadIdx.x & 31;
    const int grp  = lane >> 4;
    const int nl   = lane & 15;

    v8f acc0 = {}, acc1 = {}, acc2 = {}, acc3 = {};

    const _Float16* wp = Wp + ((size_t)rt * KT * 32 + lane) * 16;
    const _Float16* x0 = Xp + (((size_t)b * KT) * NPTS + n0 + nl) * 32 + grp * 16;

    for (int kt = 0; kt < KT; ++kt) {
        v16h a = *(const v16h*)wp;
        wp += 32 * 16;
        const _Float16* xk = x0 + (size_t)kt * NPTS * 32;
        v16h b0 = *(const v16h*)(xk);
        v16h b1 = *(const v16h*)(xk + 16 * 32);
        v16h b2 = *(const v16h*)(xk + 32 * 32);
        v16h b3 = *(const v16h*)(xk + 48 * 32);
        acc0 = __builtin_amdgcn_wmma_f32_16x16x32_f16(false, a, false, b0, (short)0, acc0, false, false);
        acc1 = __builtin_amdgcn_wmma_f32_16x16x32_f16(false, a, false, b1, (short)0, acc1, false, false);
        acc2 = __builtin_amdgcn_wmma_f32_16x16x32_f16(false, a, false, b2, (short)0, acc2, false, false);
        acc3 = __builtin_amdgcn_wmma_f32_16x16x32_f16(false, a, false, b3, (short)0, acc3, false, false);
    }

    float* Yb = Y + (long)b * ybst;
#pragma unroll
    for (int t = 0; t < 4; ++t) {
        const v8f& acc = (t == 0) ? acc0 : (t == 1) ? acc1 : (t == 2) ? acc2 : acc3;
        const int col = n0 + t * 16 + nl;
#pragma unroll
        for (int r = 0; r < 8; ++r) {
            int o = rt * 16 + r + 8 * grp;         // C/D layout: VGPR r -> M=r+8*grp
            if (o < O) {
                float v = acc[r];
                if constexpr (EPI == 1) { v += abias[b * O + o]; }
                if constexpr (EPI == 2) {
                    if (abias) v += abias[b * O + o];
                    v = lrelu_f(v * (g[o] * INVBN) + bb[o]);
                }
                if constexpr (EPI == 3) {
                    v = v * (g[o] * INVBN) + bb[o]
                        + addm[(long)b * addbst + (long)o * NPTS + col];
                    v = lrelu_f(v);
                }
                Yb[(long)o * NPTS + col] = v;
            }
        }
    }
}

// ---------------------------------------------------------------------------
// Edge-conv finish: out[b,o,n] = max_k lrelu(bn(A[b,o,idx[n,k]] + Bc[b,o,n]))
// ---------------------------------------------------------------------------
__global__ void edge_combine(const float* __restrict__ A, long abst,
                             const float* __restrict__ Bc, long bcbst,
                             const int* __restrict__ idx,
                             const float* __restrict__ g, const float* __restrict__ bb,
                             float* __restrict__ out, long obst, int O)
{
    long t = (long)blockIdx.x * blockDim.x + threadIdx.x;
    if (t >= (long)NB * O * NPTS) return;
    int n = (int)(t % NPTS);
    int o = (int)((t / NPTS) % O);
    int b = (int)(t / ((long)O * NPTS));
    const float* Ar = A + (long)b * abst + (long)o * NPTS;
    float bc = Bc[(long)b * bcbst + (long)o * NPTS + n];
    float s = g[o] * INVBN, bo = bb[o];
    const int* ip = idx + ((long)b * NPTS + n) * KNN;
    float m = -3.4e38f;
#pragma unroll
    for (int k = 0; k < KNN; ++k) {
        float v = lrelu_f((Ar[ip[k]] + bc) * s + bo);
        m = fmaxf(m, v);
    }
    out[(long)b * obst + (long)o * NPTS + n] = m;
}

// ---------------------------------------------------------------------------
// Global max over points: glob[b,o] = max_n PE[b,o,n]   (emb rows only)
// ---------------------------------------------------------------------------
__global__ void glob_max(const float* __restrict__ PE, long pbst, float* __restrict__ glob)
{
    __shared__ float red[256];
    int o = blockIdx.x, b = blockIdx.y, tid = threadIdx.x;
    const float* p = PE + (long)b * pbst + (long)o * NPTS;
    float m = -3.4e38f;
    for (int i = tid; i < NPTS; i += 256) m = fmaxf(m, p[i]);
    red[tid] = m; __syncthreads();
    for (int s = 128; s > 0; s >>= 1) {
        if (tid < s) red[tid] = fmaxf(red[tid], red[tid + s]);
        __syncthreads();
    }
    if (tid == 0) glob[b * 512 + o] = red[0];
}

// ---------------------------------------------------------------------------
// Label branch: lblf[b,o] = lrelu(bn(w_lbl @ lbl))   (8x64)
// ---------------------------------------------------------------------------
__global__ void lbl_kernel(const float* __restrict__ w, const float* __restrict__ g,
                           const float* __restrict__ bb, const float* __restrict__ lbl,
                           float* __restrict__ lblf)
{
    int t = blockIdx.x * blockDim.x + threadIdx.x;
    if (t >= NB * 64) return;
    int b = t >> 6, o = t & 63;
    float s = 0.f;
#pragma unroll
    for (int c = 0; c < 16; ++c) s += w[o * 16 + c] * lbl[b * 16 + c];
    lblf[t] = lrelu_f(s * (g[o] * INVBN) + bb[o]);
}

// ---------------------------------------------------------------------------
// Broadcast-channel bias: out[b,o] = W[o,515:1027]@glob[b] + W[o,1027:1091]@lblf[b]
// ---------------------------------------------------------------------------
__global__ void res_bias(const float* __restrict__ W, int ldw, int O,
                         const float* __restrict__ glob, const float* __restrict__ lblf,
                         float* __restrict__ out)
{
    int t = blockIdx.x * blockDim.x + threadIdx.x;
    if (t >= NB * O) return;
    int b = t / O, o = t % O;
    const float* wr = W + (long)o * ldw;
    float s = 0.f;
    for (int c = 0; c < 512; ++c) s += wr[515 + c] * glob[b * 512 + c];
    for (int c = 0; c < 64; ++c)  s += wr[1027 + c] * lblf[b * 64 + c];
    out[t] = s;
}

// xyz -> rows 512..514 of PE (zero-copy "point" concat)
__global__ void copy_xyz(const float* __restrict__ src, float* __restrict__ PE)
{
    int t = blockIdx.x * blockDim.x + threadIdx.x;
    if (t >= NB * 3 * NPTS) return;
    int n = t % NPTS, c = (t / NPTS) % 3, b = t / (3 * NPTS);
    PE[(long)b * 515 * NPTS + (long)(512 + c) * NPTS + n] =
        src[(long)b * 3 * NPTS + (long)c * NPTS + n];
}

// ---------------------------------------------------------------------------
// Host-side helpers
// ---------------------------------------------------------------------------
static inline void launch_pack_w(const float* W, int ldw, int O, int K, int KT,
                                 _Float16* Wp, hipStream_t s)
{
    int rowTiles = (O + 15) / 16;
    int n = rowTiles * KT * 32;
    pack_w<<<(n + 255) / 256, 256, 0, s>>>(W, ldw, O, K, KT, Wp);
}

static inline void launch_pack_x(const float* X, long xbst, int K, int KT,
                                 _Float16* Xp, hipStream_t s)
{
    long n = (long)NB * KT * NPTS;
    pack_x<<<(int)((n + 255) / 256), 256, 0, s>>>(X, xbst, K, KT, Xp);
}

static inline void launch_gemm(int epi, const _Float16* Wp, const _Float16* Xp,
                               int O, int KT, float* Y, long ybst,
                               const float* abias, const float* g, const float* bb,
                               const float* addm, long addbst, hipStream_t s)
{
    int rowTiles = (O + 15) / 16;
    int totalWaves = rowTiles * NB * (NPTS / 64);
    int blocks = (totalWaves + 7) / 8;
    switch (epi) {
    case 0: gemm_wmma<0><<<blocks, 256, 0, s>>>(Wp, Xp, O, KT, Y, ybst, abias, g, bb, addm, addbst); break;
    case 1: gemm_wmma<1><<<blocks, 256, 0, s>>>(Wp, Xp, O, KT, Y, ybst, abias, g, bb, addm, addbst); break;
    case 2: gemm_wmma<2><<<blocks, 256, 0, s>>>(Wp, Xp, O, KT, Y, ybst, abias, g, bb, addm, addbst); break;
    default: gemm_wmma<3><<<blocks, 256, 0, s>>>(Wp, Xp, O, KT, Y, ybst, abias, g, bb, addm, addbst); break;
    }
}

extern "C" void kernel_launch(void* const* d_in, const int* in_sizes, int n_in,
                              void* d_out, int out_size, void* d_ws, size_t ws_size,
                              hipStream_t stream)
{
    (void)in_sizes; (void)n_in; (void)out_size; (void)ws_size;
    const float* src   = (const float*)d_in[0];
    const float* lbl   = (const float*)d_in[1];
    const float* w_ec1 = (const float*)d_in[2];
    const float* g_ec1 = (const float*)d_in[3];
    const float* b_ec1 = (const float*)d_in[4];
    const float* w_ec2 = (const float*)d_in[5];
    const float* g_ec2 = (const float*)d_in[6];
    const float* b_ec2 = (const float*)d_in[7];
    const float* w_ec3 = (const float*)d_in[8];
    const float* g_ec3 = (const float*)d_in[9];
    const float* b_ec3 = (const float*)d_in[10];
    const float* w_emb = (const float*)d_in[11];
    const float* g_emb = (const float*)d_in[12];
    const float* b_emb = (const float*)d_in[13];
    const float* w_lbl = (const float*)d_in[14];
    const float* g_lbl = (const float*)d_in[15];
    const float* b_lbl = (const float*)d_in[16];
    const float* w_r1a = (const float*)d_in[17];
    const float* g_r1a = (const float*)d_in[18];
    const float* b_r1a = (const float*)d_in[19];
    const float* w_r1b = (const float*)d_in[20];
    const float* g_r1b = (const float*)d_in[21];
    const float* b_r1b = (const float*)d_in[22];
    const float* w_r1s = (const float*)d_in[23];
    const float* w_r2a = (const float*)d_in[24];
    const float* g_r2a = (const float*)d_in[25];
    const float* b_r2a = (const float*)d_in[26];
    const float* w_r2b = (const float*)d_in[27];
    const float* g_r2b = (const float*)d_in[28];
    const float* b_r2b = (const float*)d_in[29];
    const float* w_r2s = (const float*)d_in[30];

    char* ws = (char*)d_ws;
    size_t off = 0;
    auto take = [&](size_t bytes) -> char* {
        char* p = ws + off;
        off += (bytes + 255) & ~(size_t)255;
        return p;
    };
    float* XIN  = (float*)take((size_t)NB * 21 * NPTS * 4);
    int*   IDX  = (int*)  take((size_t)NB * NPTS * KNN * 4);
    float* W1B  = (float*)take(128 * 64 * 4);
    float* WDB  = (float*)take(128 * 64 * 4);
    float* ABUF = (float*)take((size_t)NB * 128 * NPTS * 4);
    float* BCB  = (float*)take((size_t)NB * 128 * NPTS * 4);
    float* CAT  = (float*)take((size_t)NB * 256 * NPTS * 4);   // x1|x2|x3
    float* PE   = (float*)take((size_t)NB * 515 * NPTS * 4);   // emb|xyz
    float* GLOB = (float*)take((size_t)NB * 512 * 4);
    float* LBLF = (float*)take((size_t)NB * 64 * 4);
    float* B1A  = (float*)take((size_t)NB * 512 * 4);
    float* B1S  = (float*)take((size_t)NB * 256 * 4);
    float* H1   = (float*)take((size_t)NB * 512 * NPTS * 4);
    float* SC   = (float*)take((size_t)NB * 256 * NPTS * 4);
    float* R1   = (float*)take((size_t)NB * 256 * NPTS * 4);
    float* H2   = (float*)take((size_t)NB * 128 * NPTS * 4);
    float* SC2  = (float*)take((size_t)NB * 50 * NPTS * 4);
    _Float16* WP1 = (_Float16*)take((size_t)32 * 17 * 32 * 16 * 2);   // max rowTiles*KT
    _Float16* WP2 = (_Float16*)take((size_t)32 * 17 * 32 * 16 * 2);
    _Float16* XP  = (_Float16*)take((size_t)NB * 17 * NPTS * 32 * 2); // max KT=17
    float* OUT  = (float*)d_out;

    const dim3 kgrid(NPTS / 128, NB);
    const long S21 = (long)21 * NPTS, S64 = (long)64 * NPTS, S128 = (long)128 * NPTS;
    const long S256 = (long)256 * NPTS, S515 = (long)515 * NPTS, S512 = (long)512 * NPTS;
    const long S50 = (long)50 * NPTS;

    // --- HOG features (knn on xyz) ---
    knn_kernel<3><<<kgrid, 128, 0, stream>>>(src, (long)3 * NPTS, IDX);
    hog_kernel<<<(NB * NPTS + 255) / 256, 256, 0, stream>>>(src, IDX, XIN);

    // --- edge conv 1 (C=21 -> O=64) -> CAT rows 0..63 ---
    knn_kernel<21><<<kgrid, 128, 0, stream>>>(XIN, S21, IDX);
    prep_w_ec<<<(64 * 21 + 255) / 256, 256, 0, stream>>>(w_ec1, 64, 21, W1B, WDB);
    launch_pack_w(W1B, 21, 64, 21, 1, WP1, stream);
    launch_pack_w(WDB, 21, 64, 21, 1, WP2, stream);
    launch_pack_x(XIN, S21, 21, 1, XP, stream);
    launch_gemm(0, WP1, XP, 64, 1, ABUF, S64, nullptr, nullptr, nullptr, nullptr, 0, stream);
    launch_gemm(0, WP2, XP, 64, 1, BCB,  S64, nullptr, nullptr, nullptr, nullptr, 0, stream);
    edge_combine<<<(NB * 64 * NPTS + 255) / 256, 256, 0, stream>>>(
        ABUF, S64, BCB, S64, IDX, g_ec1, b_ec1, CAT, S256, 64);

    // --- edge conv 2 (C=64 -> O=64) -> CAT rows 64..127 ---
    knn_kernel<64><<<kgrid, 128, 0, stream>>>(CAT, S256, IDX);
    prep_w_ec<<<(64 * 64 + 255) / 256, 256, 0, stream>>>(w_ec2, 64, 64, W1B, WDB);
    launch_pack_w(W1B, 64, 64, 64, 2, WP1, stream);
    launch_pack_w(WDB, 64, 64, 64, 2, WP2, stream);
    launch_pack_x(CAT, S256, 64, 2, XP, stream);
    launch_gemm(0, WP1, XP, 64, 2, ABUF, S64, nullptr, nullptr, nullptr, nullptr, 0, stream);
    launch_gemm(0, WP2, XP, 64, 2, BCB,  S64, nullptr, nullptr, nullptr, nullptr, 0, stream);
    edge_combine<<<(NB * 64 * NPTS + 255) / 256, 256, 0, stream>>>(
        ABUF, S64, BCB, S64, IDX, g_ec2, b_ec2, CAT + S64, S256, 64);

    // --- edge conv 3 (C=64 -> O=128) -> CAT rows 128..255 ---
    knn_kernel<64><<<kgrid, 128, 0, stream>>>(CAT + S64, S256, IDX);
    prep_w_ec<<<(128 * 64 + 255) / 256, 256, 0, stream>>>(w_ec3, 128, 64, W1B, WDB);
    launch_pack_w(W1B, 64, 128, 64, 2, WP1, stream);
    launch_pack_w(WDB, 64, 128, 64, 2, WP2, stream);
    launch_pack_x(CAT + S64, S256, 64, 2, XP, stream);
    launch_gemm(0, WP1, XP, 128, 2, ABUF, S128, nullptr, nullptr, nullptr, nullptr, 0, stream);
    launch_gemm(0, WP2, XP, 128, 2, BCB,  S128, nullptr, nullptr, nullptr, nullptr, 0, stream);
    edge_combine<<<(NB * 128 * NPTS + 255) / 256, 256, 0, stream>>>(
        ABUF, S128, BCB, S128, IDX, g_ec3, b_ec3, CAT + S128, S256, 128);

    // --- embedding GEMM (512x256) + bn + lrelu -> PE rows 0..511 ---
    launch_pack_w(w_emb, 256, 512, 256, 8, WP1, stream);
    launch_pack_x(CAT, S256, 256, 8, XP, stream);
    launch_gemm(2, WP1, XP, 512, 8, PE, S515, nullptr, g_emb, b_emb, nullptr, 0, stream);
    copy_xyz<<<(NB * 3 * NPTS + 255) / 256, 256, 0, stream>>>(src, PE);

    // --- global max + label branch + broadcast-channel biases ---
    glob_max<<<dim3(512, NB), 256, 0, stream>>>(PE, S515, GLOB);
    lbl_kernel<<<(NB * 64 + 255) / 256, 256, 0, stream>>>(w_lbl, g_lbl, b_lbl, lbl, LBLF);
    res_bias<<<(NB * 512 + 255) / 256, 256, 0, stream>>>(w_r1a, 1091, 512, GLOB, LBLF, B1A);
    res_bias<<<(NB * 256 + 255) / 256, 256, 0, stream>>>(w_r1s, 1091, 256, GLOB, LBLF, B1S);

    // --- residual layer 1 (K=515, KT=17) ---
    launch_pack_x(PE, S515, 515, 17, XP, stream);
    launch_pack_w(w_r1a, 1091, 512, 515, 17, WP1, stream);
    launch_pack_w(w_r1s, 1091, 256, 515, 17, WP2, stream);
    launch_gemm(2, WP1, XP, 512, 17, H1, S512, B1A, g_r1a, b_r1a, nullptr, 0, stream);
    launch_gemm(1, WP2, XP, 256, 17, SC, S256, B1S, nullptr, nullptr, nullptr, 0, stream);
    launch_pack_x(H1, S512, 512, 16, XP, stream);
    launch_pack_w(w_r1b, 512, 256, 512, 16, WP1, stream);
    launch_gemm(3, WP1, XP, 256, 16, R1, S256, nullptr, g_r1b, b_r1b, SC, S256, stream);

    // --- residual layer 2 -> d_out (8,50,2048) ---
    launch_pack_x(R1, S256, 256, 8, XP, stream);
    launch_pack_w(w_r2a, 256, 128, 256, 8, WP1, stream);
    launch_pack_w(w_r2s, 256, 50, 256, 8, WP2, stream);
    launch_gemm(2, WP1, XP, 128, 8, H2, S128, nullptr, g_r2a, b_r2a, nullptr, 0, stream);
    launch_gemm(0, WP2, XP, 50, 8, SC2, S50, nullptr, nullptr, nullptr, nullptr, 0, stream);
    launch_pack_x(H2, S128, 128, 4, XP, stream);
    launch_pack_w(w_r2b, 128, 50, 128, 4, WP1, stream);
    launch_gemm(3, WP1, XP, 50, 4, OUT, S50, nullptr, g_r2b, b_r2b, SC2, S50, stream);
}
